// TemporalCfCEncoder_22737556865384
// MI455X (gfx1250) — matrix operand order repaired and checked by reading, not compiled, not run on hardware
//
#include <hip/hip_runtime.h>

#define T_  512
#define N_  256
#define D_  256
#define H_  4
#define DH_ 64
#define L_  2

typedef __attribute__((ext_vector_type(16))) __bf16   bf16x16;
typedef __attribute__((ext_vector_type(8)))  float    f32x8;
typedef __attribute__((ext_vector_type(4)))  unsigned u32x4;

union Frag { u32x4 q[2]; bf16x16 v; };

__device__ __forceinline__ unsigned short f2bf(float f) {
    unsigned u = __float_as_uint(f);
    u += 0x7FFFu + ((u >> 16) & 1u);          // round-to-nearest-even
    return (unsigned short)(u >> 16);
}

__device__ __forceinline__ f32x8 wmma_bf16(const Frag& a, const Frag& b, f32x8 c) {
    return __builtin_amdgcn_wmma_f32_16x16x32_bf16(false, a.v, false, b.v,
                                                   (short)0, c, false, false);
}

// ---------------------------------------------------------------------------
// prep: decay[l][t][d] = exp(-dt[t]/tau[l][d]), re_bias[d]
// grid = L*T + 1 blocks of 256
// ---------------------------------------------------------------------------
__global__ __launch_bounds__(256) void prep_kernel(
    const float* __restrict__ sensor_time, const float* __restrict__ log_tau,
    const float* __restrict__ re_norm, const float* __restrict__ re_w,
    const float* __restrict__ re_b, float* __restrict__ decay,
    float* __restrict__ rebias)
{
    int b = blockIdx.x, tid = threadIdx.x;
    if (b < L_ * T_) {
        int l = b >> 9, t = b & (T_ - 1);
        float dt = (t == 0) ? sensor_time[0] : sensor_time[t] - sensor_time[t - 1];
        float lt = log_tau[l * D_ + tid];
        lt = fminf(1.f, fmaxf(-1.f, lt));
        float tau = __expf(lt);
        decay[((size_t)l * T_ + t) * D_ + tid] = __expf(-dt / tau);
    } else {
        rebias[tid] = re_norm[0] * re_w[tid] + re_b[tid];
    }
}

// ---------------------------------------------------------------------------
// pack weight f32 [256(k)][256(n)] -> WMMA B-fragment layout bf16
// dst[((no*8+ko)*32+lane)*16 + e] = W[ko*32 + (e/8)*16 + (lane/16)*8 + e%8][no*16 + lane%16]
// grid = 128 (no*8+ko), block = 32
// ---------------------------------------------------------------------------
__global__ __launch_bounds__(32) void pack_w_kernel(
    const float* __restrict__ W, unsigned short* __restrict__ dst)
{
    int tile = blockIdx.x;           // no*8 + ko
    int lane = threadIdx.x;
    int no = tile >> 3, ko = tile & 7;
    int n  = no * 16 + (lane & 15);
    int kb = ko * 32 + (lane >> 4) * 8;
    unsigned short* d = dst + ((size_t)tile * 32 + lane) * 16;
#pragma unroll
    for (int e = 0; e < 16; ++e)
        d[e] = f2bf(W[(size_t)(kb + (e >> 3) * 16 + (e & 7)) * D_ + n]);
}

// ---------------------------------------------------------------------------
// LayerNorm over D=256: one row per 256-thread block, output bf16
// ---------------------------------------------------------------------------
__global__ __launch_bounds__(256) void ln_kernel(
    const float* __restrict__ x, const float* __restrict__ gw,
    const float* __restrict__ bw, unsigned short* __restrict__ out)
{
    size_t row = blockIdx.x;
    int tid = threadIdx.x, lane = tid & 31, wave = tid >> 5;
    float v = x[row * D_ + tid];
    float s = v, s2 = v * v;
#pragma unroll
    for (int o = 16; o > 0; o >>= 1) {
        s  += __shfl_down(s,  o);
        s2 += __shfl_down(s2, o);
    }
    __shared__ float sred[8], s2red[8];
    if (lane == 0) { sred[wave] = s; s2red[wave] = s2; }
    __syncthreads();
    float ts = 0.f, ts2 = 0.f;
#pragma unroll
    for (int i = 0; i < 8; ++i) { ts += sred[i]; ts2 += s2red[i]; }
    float mean = ts * (1.f / 256.f);
    float var  = ts2 * (1.f / 256.f) - mean * mean;
    float xn = (v - mean) * rsqrtf(var + 1e-5f) * gw[tid] + bw[tid];
    out[row * D_ + tid] = f2bf(xn);
}

// ---------------------------------------------------------------------------
// GEMM: C[M,256] = A_bf16[M,256] @ Wpacked + bias (+res +rebias)
// out as bf16 and/or f32.  block = 256 (8 waves), wave -> 16 rows x 64 cols
// grid = (256/64, M/128)
// ---------------------------------------------------------------------------
__global__ __launch_bounds__(256) void gemm256_kernel(
    const unsigned short* __restrict__ A, const unsigned short* __restrict__ Wp,
    const float* __restrict__ bias, const float* __restrict__ res,
    const float* __restrict__ rebias, unsigned short* __restrict__ outb,
    float* __restrict__ outf)
{
    int tid = threadIdx.x, wave = tid >> 5, lane = tid & 31;
    int g = lane >> 4, ln16 = lane & 15;
    int rowbase = blockIdx.y * 128 + wave * 16;
    int colbase = blockIdx.x * 64;

    f32x8 acc[4];
    f32x8 z = {0.f,0.f,0.f,0.f,0.f,0.f,0.f,0.f};
#pragma unroll
    for (int j = 0; j < 4; ++j) acc[j] = z;

    const unsigned short* arow = A + (size_t)(rowbase + ln16) * D_;
#pragma unroll
    for (int kk = 0; kk < 8; ++kk) {
        Frag a;
        const u32x4* ap = (const u32x4*)(arow + kk * 32 + g * 8);
        a.q[0] = ap[0];
        a.q[1] = ap[2];                       // +16 elements
#pragma unroll
        for (int j = 0; j < 4; ++j) {
            Frag b;
            const u32x4* bp = (const u32x4*)(Wp +
                (((size_t)((colbase >> 4) + j) * 8 + kk) * 32 + lane) * 16);
            b.q[0] = bp[0]; b.q[1] = bp[1];
            acc[j] = wmma_bf16(a, b, acc[j]);
        }
    }

    int moff = g * 8;
#pragma unroll
    for (int j = 0; j < 4; ++j) {
        int n = colbase + j * 16 + ln16;
        float bn = bias ? bias[n] : 0.f;
        if (rebias) bn += rebias[n];
#pragma unroll
        for (int r = 0; r < 8; ++r) {
            int row = rowbase + moff + r;
            size_t off = (size_t)row * D_ + n;
            float val = acc[j][r] + bn;
            if (res)  val += res[off];
            if (outf) outf[off] = val;
            if (outb) outb[off] = f2bf(val);
        }
    }
}

// ---------------------------------------------------------------------------
// Attention: one block per (t, h, qblock-of-64).  block = 256 (8 waves)
// ---------------------------------------------------------------------------
#define SP_ 264     // padded LDS row stride (elements)

__global__ __launch_bounds__(256) void attn_kernel(
    const unsigned short* __restrict__ q, const unsigned short* __restrict__ k,
    const unsigned short* __restrict__ v, unsigned short* __restrict__ out)
{
    const int t = blockIdx.x, h = blockIdx.y, qb = blockIdx.z;
    const int tid = threadIdx.x, wave = tid >> 5, lane = tid & 31;
    const int g = lane >> 4, ln16 = lane & 15;

    __shared__ __align__(16) float          S[64 * SP_];
    __shared__ __align__(16) unsigned short P[64 * SP_];
    __shared__ __align__(16) unsigned short VT[64 * SP_];

    const size_t baseT = ((size_t)t * N_) * D_ + h * DH_;

    // stage v^T into LDS: VT[feature][token]
    for (int idx = tid; idx < N_ * DH_; idx += 256) {
        int tok = idx >> 6, d = idx & 63;
        VT[d * SP_ + tok] = v[baseT + (size_t)tok * D_ + d];
    }

    // ---- scores: S[64 x 256] = q_block @ k^T (per wave: 1 qtile x 8 ktiles)
    int qt  = wave & 3;
    int ks0 = (wave >> 2) * 8;
    int qrow = qb * 64 + qt * 16 + ln16;
    Frag a0, a1;
    {
        const u32x4* ap = (const u32x4*)(q + baseT + (size_t)qrow * D_ + g * 8);
        a0.q[0] = ap[0]; a0.q[1] = ap[2];     // K 0..31
        a1.q[0] = ap[4]; a1.q[1] = ap[6];     // K 32..63
    }
#pragma unroll
    for (int kt = 0; kt < 8; ++kt) {
        int ktile = ks0 + kt;
        const u32x4* bp = (const u32x4*)(k + baseT +
                          (size_t)(ktile * 16 + ln16) * D_ + g * 8);
        Frag b0, b1;
        b0.q[0] = bp[0]; b0.q[1] = bp[2];     // K 0..31
        b1.q[0] = bp[4]; b1.q[1] = bp[6];     // K 32..63
        f32x8 acc = {0.f,0.f,0.f,0.f,0.f,0.f,0.f,0.f};
        acc = wmma_bf16(a0, b0, acc);
        acc = wmma_bf16(a1, b1, acc);
        int srow = qt * 16 + g * 8;
        int scol = ktile * 16 + ln16;
#pragma unroll
        for (int r = 0; r < 8; ++r) S[(srow + r) * SP_ + scol] = acc[r];
    }
    __syncthreads();

    // ---- softmax: 4 threads per row, shuffle-reduce
    {
        const float scale = 0.125f;           // 1/sqrt(64)
        int r = tid >> 2, sub = tid & 3;
        float* Sr = &S[r * SP_];
        int c0 = sub * 64;
        float mx = -1e30f;
        for (int c = c0; c < c0 + 64; ++c) mx = fmaxf(mx, Sr[c] * scale);
        mx = fmaxf(mx, __shfl_xor(mx, 1));
        mx = fmaxf(mx, __shfl_xor(mx, 2));
        float sum = 0.f;
        for (int c = c0; c < c0 + 64; ++c) {
            float e = __expf(Sr[c] * scale - mx);
            Sr[c] = e; sum += e;
        }
        sum += __shfl_xor(sum, 1);
        sum += __shfl_xor(sum, 2);
        float inv = 1.f / sum;
        for (int c = c0; c < c0 + 64; ++c) P[r * SP_ + c] = f2bf(Sr[c] * inv);
    }
    __syncthreads();

    // ---- out[64 x 64] = P @ v   (16 tiles, 2 per wave)
#pragma unroll
    for (int tt = 0; tt < 2; ++tt) {
        int id = wave + tt * 8;
        int qi = id >> 2, di = id & 3;
        f32x8 acc = {0.f,0.f,0.f,0.f,0.f,0.f,0.f,0.f};
        const u32x4* pap = (const u32x4*)(&P[(qi * 16 + ln16) * SP_]);
        const u32x4* vbp = (const u32x4*)(&VT[(di * 16 + ln16) * SP_]);
#pragma unroll
        for (int kk = 0; kk < 8; ++kk) {
            Frag a, b;
            a.q[0] = pap[kk * 4 + g];
            a.q[1] = pap[kk * 4 + g + 2];
            b.q[0] = vbp[kk * 4 + g];
            b.q[1] = vbp[kk * 4 + g + 2];
            acc = wmma_bf16(a, b, acc);
        }
        int orow = qb * 64 + qi * 16 + g * 8;
        int ocol = di * 16 + ln16;
#pragma unroll
        for (int r = 0; r < 8; ++r)
            out[baseT + (size_t)(orow + r) * D_ + ocol] = f2bf(acc[r]);
    }
}

// ---------------------------------------------------------------------------
// scan init: zero h state + barrier counter.  grid = 256, block = 256
// ---------------------------------------------------------------------------
__global__ __launch_bounds__(256) void initscan_kernel(
    float* hf, unsigned short* hb0, unsigned short* hb1, unsigned* cnt)
{
    int idx = blockIdx.x * 256 + threadIdx.x;
    hf[idx] = 0.f; hb0[idx] = 0; hb1[idx] = 0;
    if (idx == 0) *cnt = 0u;
}

// ---------------------------------------------------------------------------
// persistent CfC scan: 512 steps, per step h@wh (WMMA) + elementwise update.
// grid = 16 blocks x 256 threads (all resident); atomic-counter global barrier.
// ---------------------------------------------------------------------------
__global__ __launch_bounds__(256) void cfc_scan_kernel(
    const float* __restrict__ xproj, const float* __restrict__ decay_l,
    const float* __restrict__ seq_in, const unsigned short* __restrict__ whp,
    float* __restrict__ out, float* __restrict__ hf,
    unsigned short* hb0, unsigned short* hb1, unsigned* cnt)
{
    int tid = threadIdx.x, wave = tid >> 5, lane = tid & 31;
    int g = lane >> 4, ln16 = lane & 15;
    int rowbase = blockIdx.x * 16;
    int colbase = wave * 32;              // two 16-wide tiles per wave
    unsigned short* cur = hb0;
    unsigned short* nxt = hb1;

    for (int t = 0; t < T_; ++t) {
        f32x8 acc0 = {0.f,0.f,0.f,0.f,0.f,0.f,0.f,0.f};
        f32x8 acc1 = acc0;
        const unsigned short* arow = cur + (size_t)(rowbase + ln16) * D_;
#pragma unroll
        for (int kk = 0; kk < 8; ++kk) {
            Frag a;
            const u32x4* ap = (const u32x4*)(arow + kk * 32 + g * 8);
            a.q[0] = ap[0]; a.q[1] = ap[2];
            Frag b0, b1;
            const u32x4* bp0 = (const u32x4*)(whp +
                (((size_t)(colbase >> 4) * 8 + kk) * 32 + lane) * 16);
            const u32x4* bp1 = (const u32x4*)(whp +
                (((size_t)((colbase >> 4) + 1) * 8 + kk) * 32 + lane) * 16);
            b0.q[0] = bp0[0]; b0.q[1] = bp0[1];
            b1.q[0] = bp1[0]; b1.q[1] = bp1[1];
            acc0 = wmma_bf16(a, b0, acc0);
            acc1 = wmma_bf16(a, b1, acc1);
        }
        size_t tb = (size_t)t * (N_ * D_);
#pragma unroll
        for (int j = 0; j < 2; ++j) {
            int n = colbase + j * 16 + ln16;
            float dec = decay_l[t * D_ + n];
#pragma unroll
            for (int r = 0; r < 8; ++r) {
                int row = rowbase + g * 8 + r;
                size_t off = (size_t)row * D_ + n;
                float hw   = j ? acc1[r] : acc0[r];
                float xp   = xproj[tb + off];
                float hold = hf[off];
                float hn = hold * dec + (1.f - dec) * tanhf(xp + hw);
                hf[off]  = hn;
                nxt[off] = f2bf(hn);
                out[tb + off] = seq_in ? hn + seq_in[tb + off] : hn;
            }
        }
        // ---- global barrier across the 16 resident workgroups
        __syncthreads();
        if (tid == 0) {
            __threadfence();
            atomicAdd(cnt, 1u);
            unsigned target = 16u * (unsigned)(t + 1);
            while (atomicAdd(cnt, 0u) < target) { }
        }
        __syncthreads();
        __threadfence();
        unsigned short* tmp = cur; cur = nxt; nxt = tmp;
    }
}

// ---------------------------------------------------------------------------
extern "C" void kernel_launch(void* const* d_in, const int* in_sizes, int n_in,
                              void* d_out, int out_size, void* d_ws, size_t ws_size,
                              hipStream_t stream) {
    const float* spatial  = (const float*)d_in[0];
    const float* re_norm  = (const float*)d_in[1];
    const float* sensor_t = (const float*)d_in[2];
    const float* re_w     = (const float*)d_in[3];
    const float* re_b     = (const float*)d_in[4];
    const float* ln_g     = (const float*)d_in[5];
    const float* ln_b     = (const float*)d_in[6];
    const float* wq       = (const float*)d_in[7];
    const float* wk       = (const float*)d_in[8];
    const float* wv       = (const float*)d_in[9];
    const float* wo       = (const float*)d_in[10];
    const float* bq       = (const float*)d_in[11];
    const float* bk       = (const float*)d_in[12];
    const float* bv       = (const float*)d_in[13];
    const float* bo       = (const float*)d_in[14];
    const float* cell_wx  = (const float*)d_in[15];
    const float* cell_wh  = (const float*)d_in[16];
    const float* cell_b   = (const float*)d_in[17];
    const float* log_tau  = (const float*)d_in[18];
    float* out = (float*)d_out;

    const size_t SZF = (size_t)T_ * N_ * D_ * 4;   // 134 MB f32
    const size_t SZH = SZF / 2;                    // 67 MB bf16
    char* w = (char*)d_ws;

    float*          seqws  = (float*)w;                                    // layer0 out
    unsigned short* xn     = (unsigned short*)(w + SZF);                   // LN out / attn_out alias
    unsigned short* qbuf   = (unsigned short*)(w + SZF + SZH);             // q / x_bf16 alias
    unsigned short* kbuf   = (unsigned short*)(w + SZF + 2 * SZH);
    unsigned short* vbuf   = (unsigned short*)(w + SZF + 3 * SZH);
    float*          xproj  = (float*)(w + SZF + 2 * SZH);                  // alias k+v
    unsigned short* attn_o = xn;
    unsigned short* xbf    = qbuf;

    char* sm = w + SZF + 4 * SZH;
    float*          decay  = (float*)sm;                               sm += (size_t)L_ * T_ * D_ * 4;
    float*          rebias = (float*)sm;                               sm += 1024;
    unsigned short* packs  = (unsigned short*)sm;                      sm += (size_t)8 * D_ * D_ * 2;
    float*          hf     = (float*)sm;                               sm += (size_t)N_ * D_ * 4;
    unsigned short* hb0    = (unsigned short*)sm;                      sm += (size_t)N_ * D_ * 2;
    unsigned short* hb1    = (unsigned short*)sm;                      sm += (size_t)N_ * D_ * 2;
    unsigned*       cnt    = (unsigned*)sm;

    unsigned short* wq_p  = packs + 0 * 65536;
    unsigned short* wk_p  = packs + 1 * 65536;
    unsigned short* wv_p  = packs + 2 * 65536;
    unsigned short* wo_p  = packs + 3 * 65536;
    unsigned short* wx_p0 = packs + 4 * 65536;
    unsigned short* wx_p1 = packs + 5 * 65536;
    unsigned short* wh_p0 = packs + 6 * 65536;
    unsigned short* wh_p1 = packs + 7 * 65536;

    prep_kernel<<<L_ * T_ + 1, 256, 0, stream>>>(sensor_t, log_tau, re_norm,
                                                 re_w, re_b, decay, rebias);
    pack_w_kernel<<<128, 32, 0, stream>>>(wq, wq_p);
    pack_w_kernel<<<128, 32, 0, stream>>>(wk, wk_p);
    pack_w_kernel<<<128, 32, 0, stream>>>(wv, wv_p);
    pack_w_kernel<<<128, 32, 0, stream>>>(wo, wo_p);
    pack_w_kernel<<<128, 32, 0, stream>>>(cell_wx,          wx_p0);
    pack_w_kernel<<<128, 32, 0, stream>>>(cell_wx + 65536,  wx_p1);
    pack_w_kernel<<<128, 32, 0, stream>>>(cell_wh,          wh_p0);
    pack_w_kernel<<<128, 32, 0, stream>>>(cell_wh + 65536,  wh_p1);

    dim3 gGem(4, 1024, 1);
    for (int l = 0; l < L_; ++l) {
        const float* seq_prev = (l == 0) ? spatial : seqws;
        float* seq_out = (l == L_ - 1) ? out : seqws;
        unsigned short* wx_p = (l == 0) ? wx_p0 : wx_p1;
        unsigned short* wh_p = (l == 0) ? wh_p0 : wh_p1;

        ln_kernel<<<T_ * N_, 256, 0, stream>>>(seq_prev, ln_g, ln_b, xn);

        gemm256_kernel<<<gGem, 256, 0, stream>>>(xn, wq_p, bq, nullptr, nullptr, qbuf, nullptr);
        gemm256_kernel<<<gGem, 256, 0, stream>>>(xn, wk_p, bk, nullptr, nullptr, kbuf, nullptr);
        gemm256_kernel<<<gGem, 256, 0, stream>>>(xn, wv_p, bv, nullptr, nullptr, vbuf, nullptr);

        attn_kernel<<<dim3(T_, H_, 4), 256, 0, stream>>>(qbuf, kbuf, vbuf, attn_o);

        // x = seq + attn_out@wo + bo + re_bias  -> bf16
        gemm256_kernel<<<gGem, 256, 0, stream>>>(attn_o, wo_p, bo, seq_prev, rebias,
                                                 xbf, nullptr);
        // xproj = x @ wx + b  -> f32
        gemm256_kernel<<<gGem, 256, 0, stream>>>(xbf, wx_p, cell_b + l * D_,
                                                 nullptr, nullptr, nullptr, xproj);

        initscan_kernel<<<256, 256, 0, stream>>>(hf, hb0, hb1, cnt);
        cfc_scan_kernel<<<16, 256, 0, stream>>>(xproj, decay + (size_t)l * T_ * D_,
                                                (l > 0) ? seqws : nullptr, wh_p,
                                                seq_out, hf, hb0, hb1, cnt);
    }
}
